// ANA_37409165148290
// MI455X (gfx1250) — compile-verified, matrix operand order checked by reading, not run on hardware
//
#include <hip/hip_runtime.h>

// ---------------------------------------------------------------------------
// Problem constants (match reference)
// ---------------------------------------------------------------------------
#define NB      64      // batch
#define NN      128     // nodes per graph
#define NE      512     // edges per graph
#define FN      16      // node features
#define FE      8       // edge features
#define HS      64      // NODE_SIZE == EDGE_SIZE
#define NPAIR   (NN*NN) // 16384
#define EPSC    1e-6f

typedef __attribute__((ext_vector_type(16))) _Float16 v16h;
typedef __attribute__((ext_vector_type(8)))  _Float16 h8;
typedef __attribute__((ext_vector_type(8)))  float    v8f;

// ---------------------------------------------------------------------------
// Activation helpers
// ---------------------------------------------------------------------------
__device__ __forceinline__ float softplusf(float x) {
  return (x > 15.f) ? x : __logf(1.f + __expf(x));
}
__device__ __forceinline__ float milaf(float x) {
  return x * tanhf(softplusf(x - 1.f));
}
__device__ __forceinline__ float sspf(float x) {
  return softplusf(x) + 0.001f;
}

// ---------------------------------------------------------------------------
// Weight f32 -> f16 conversion + fragment-major packing.
// Packed layout: Wp[((frag*32)+lane)*16 + e], frag = kt*4+nt,
// element e=2v+p -> (k = kt*32 + (v>>2)*16 + (lane>>4)*8 + (v&3)*2 + p,
//                    n = nt*16 + (lane&15))
// so a B-fragment is one contiguous 32B per lane.
// ---------------------------------------------------------------------------
__global__ void convert_pack_kernel(const float* __restrict__ W0,
                                    const float* __restrict__ W1,
                                    _Float16* __restrict__ W0p,
                                    _Float16* __restrict__ W1p) {
  int idx = blockIdx.x * 256 + threadIdx.x;
  if (idx < 16 * 32 * 16) {  // W0: 16 fragments (kt 0..3, nt 0..3)
    int e = idx & 15, lane = (idx >> 4) & 31, frag = idx >> 9;
    int kt = frag >> 2, nt = frag & 3;
    int v = e >> 1, p = e & 1;
    int k = kt * 32 + (v >> 2) * 16 + (lane >> 4) * 8 + (v & 3) * 2 + p;
    int n = nt * 16 + (lane & 15);
    W0p[idx] = (_Float16)W0[k * 64 + n];
  }
  if (idx < 8 * 32 * 16) {   // W1: 8 fragments (kt 0..1, nt 0..3)
    int e = idx & 15, lane = (idx >> 4) & 31, frag = idx >> 9;
    int kt = frag >> 2, nt = frag & 3;
    int v = e >> 1, p = e & 1;
    int k = kt * 32 + (v >> 2) * 16 + (lane >> 4) * 8 + (v & 3) * 2 + p;
    int n = nt * 16 + (lane & 15);
    W1p[idx] = (_Float16)W1[k * 64 + n];
  }
}

// ---------------------------------------------------------------------------
// GNN kernel: one block per graph. n/agg live in LDS, e lives in global ws.
// Produces feats (f32+f16), per-node ap params (5), h flags.
// ---------------------------------------------------------------------------
__global__ __launch_bounds__(256) void gnn_kernel(
    const float* __restrict__ nodes1, const float* __restrict__ edges1,
    const int* __restrict__ send1, const int* __restrict__ recv1,
    const float* __restrict__ nodes2, const float* __restrict__ edges2,
    const int* __restrict__ send2, const int* __restrict__ recv2,
    const float* __restrict__ Wen, const float* __restrict__ ben,
    const float* __restrict__ Wee, const float* __restrict__ bee,
    const float* __restrict__ Wge, const float* __restrict__ bge,
    const float* __restrict__ Wgn, const float* __restrict__ bgn,
    const float* __restrict__ apW0, const float* __restrict__ apb0,
    const float* __restrict__ apW1, const float* __restrict__ apb1,
    const float* __restrict__ apW2, const float* __restrict__ apb2,
    float* __restrict__ feats_f32, _Float16* __restrict__ feats_f16,
    float* __restrict__ apout, int* __restrict__ hflags,
    float* __restrict__ e_buf)
{
  const int g = blockIdx.x;
  const float* nodes = g ? nodes2 : nodes1;
  const float* edges = g ? edges2 : edges1;
  const int* sidx = g ? send2 : send1;
  const int* ridx = g ? recv2 : recv1;
  float* eb = e_buf + (size_t)g * NE * HS;
  float* fout = feats_f32 + (size_t)g * NN * HS;
  _Float16* fouth = feats_f16 + (size_t)g * NN * HS;
  float* apo = apout + (size_t)g * NN * 5;
  int* hf = hflags + (size_t)g * NN;

  extern __shared__ char smem[];
  float* n_s   = (float*)smem;          // 128*64
  float* agg_s = n_s + NN * HS;         // 128*64

  const int tid = threadIdx.x;

  // node embedding
  for (int idx = tid; idx < NN * HS; idx += 256) {
    int i = idx >> 6, c = idx & 63;
    float acc = ben[c];
#pragma unroll
    for (int k = 0; k < FN; ++k) acc += nodes[i * FN + k] * Wen[k * HS + c];
    n_s[idx] = milaf(acc);
  }
  // edge embedding -> global e buffer
  for (int idx = tid; idx < NE * HS; idx += 256) {
    int e = idx >> 6, c = idx & 63;
    float acc = bee[c];
#pragma unroll
    for (int k = 0; k < FE; ++k) acc += edges[e * FE + k] * Wee[k * HS + c];
    eb[idx] = milaf(acc);
  }
  __threadfence_block();
  __syncthreads();

  for (int s = 0; s < 3; ++s) {
    const float* We = Wge + (size_t)s * 192 * HS;
    const float* be = bge + (size_t)s * HS;
    const float* Wn = Wgn + (size_t)s * 128 * HS;
    const float* bn = bgn + (size_t)s * HS;

    // edge update: 8 chunks of 64 edges; e_new[e] depends only on e[e], n
    for (int ch = 0; ch < 8; ++ch) {
      int e = ch * 64 + (tid >> 2);
      int cb = (tid & 3) * 16;
      int rr = ridx[e], ss = sidx[e];
      float tmp[16];
      for (int cc = 0; cc < 16; ++cc) {
        int c = cb + cc;
        float acc = be[c];
        for (int k = 0; k < HS; ++k) acc += eb[e * HS + k] * We[k * HS + c];
        for (int k = 0; k < HS; ++k) acc += n_s[rr * HS + k] * We[(64 + k) * HS + c];
        for (int k = 0; k < HS; ++k) acc += n_s[ss * HS + k] * We[(128 + k) * HS + c];
        tmp[cc] = milaf(acc);
      }
      __syncthreads();
      for (int cc = 0; cc < 16; ++cc) eb[e * HS + cb + cc] = tmp[cc];
      __threadfence_block();
      __syncthreads();
    }

    // segment_sum over receivers into agg_s
    for (int idx = tid; idx < NN * HS; idx += 256) agg_s[idx] = 0.f;
    __syncthreads();
    for (int idx = tid; idx < NE * HS; idx += 256) {
      int e = idx >> 6, c = idx & 63;
      atomicAdd(&agg_s[ridx[e] * HS + c], eb[idx]);
    }
    __syncthreads();

    // node update: n_new[i] depends only on n[i], agg[i]
    {
      int i = tid >> 1;
      int cb = (tid & 1) * 32;
      float tmp[32];
      for (int cc = 0; cc < 32; ++cc) {
        int c = cb + cc;
        float acc = bn[c];
        for (int k = 0; k < HS; ++k) acc += n_s[i * HS + k] * Wn[k * HS + c];
        for (int k = 0; k < HS; ++k) acc += agg_s[i * HS + k] * Wn[(64 + k) * HS + c];
        tmp[cc] = milaf(acc);
      }
      __syncthreads();
      for (int cc = 0; cc < 32; ++cc) n_s[i * HS + cb + cc] = tmp[cc];
      __syncthreads();
    }
  }

  // write feats (f32 + f16)
  for (int idx = tid; idx < NN * HS; idx += 256) {
    float v = n_s[idx];
    fout[idx] = v;
    fouth[idx] = (_Float16)v;
  }
  __syncthreads();

  // atom-param MLP: h0 -> agg_s
  for (int idx = tid; idx < NN * HS; idx += 256) {
    int i = idx >> 6, c = idx & 63;
    float acc = apb0[c];
    for (int k = 0; k < HS; ++k) acc += n_s[i * HS + k] * apW0[k * HS + c];
    agg_s[idx] = milaf(acc);
  }
  __syncthreads();
  // h1 -> n_s (n_s free now; reads agg_s only)
  for (int idx = tid; idx < NN * HS; idx += 256) {
    int i = idx >> 6, c = idx & 63;
    float acc = apb1[c];
    for (int k = 0; k < HS; ++k) acc += agg_s[i * HS + k] * apW1[k * HS + c];
    n_s[idx] = milaf(acc);
  }
  __syncthreads();
  // out5 = ssp(h1 @ W2 + b2)
  for (int idx = tid; idx < NN * 5; idx += 256) {
    int i = idx / 5, o = idx % 5;
    float acc = apb2[o];
    for (int k = 0; k < HS; ++k) acc += n_s[i * HS + k] * apW2[k * 5 + o];
    apo[idx] = sspf(acc);
  }
  // argmax(nodes,-1)==0 flags
  for (int i = tid; i < NN; i += 256) {
    float m0 = nodes[i * FN];
    int f = 1;
#pragma unroll
    for (int k = 1; k < FN; ++k) if (nodes[i * FN + k] > m0) f = 0;
    hf[i] = f;
  }
}

// ---------------------------------------------------------------------------
// WMMA fragment loads.
// A (16-bit, 16x32, ISA 7.12.2): lane's 16 halfs are two contiguous 8-half
// runs at k = kt*32 + kh*8 and kt*32 + 16 + kh*8 (runs start at multiples of
// 8 -> never cross the kmask wrap) => two ds_load_b128.
// B: loaded from fragment-major packed global weights, 32B/lane vector load.
// ---------------------------------------------------------------------------
__device__ __forceinline__ v16h load_a_frag(const _Float16* base, int stride,
                                            int kt, int koff, int kmask, int lane) {
  int row = lane & 15, kh = lane >> 4;
  int k0 = (kt * 32 + kh * 8 + koff) & kmask;
  int k1 = (kt * 32 + 16 + kh * 8 + koff) & kmask;
  h8 lo = *(const h8*)(base + row * stride + k0);
  h8 hi = *(const h8*)(base + row * stride + k1);
  v16h a;
#pragma unroll
  for (int e = 0; e < 8; ++e) { a[e] = lo[e]; a[8 + e] = hi[e]; }
  return a;
}
__device__ __forceinline__ v16h load_b_packed(const _Float16* wp, int frag, int lane) {
  return *(const v16h*)(wp + ((size_t)(frag * 32 + lane)) * 16);
}

// ---------------------------------------------------------------------------
// Pair MLP kernel: params = ssp-MLP(concat(f1,f2)) + ssp-MLP(concat(f2,f1))
// One wave per 16-pair tile; WMMA f32<-f16 for layers 0 and 1.
// 64 blocks x 128 threads; each block: 16 tiles (4 waves x 4 tiles).
// ---------------------------------------------------------------------------
__global__ __launch_bounds__(128) void pair_kernel(
    const _Float16* __restrict__ f1h, const _Float16* __restrict__ f2h,
    const _Float16* __restrict__ W0p, const _Float16* __restrict__ W1p,
    const float* __restrict__ b0, const float* __restrict__ b1,
    const float* __restrict__ W2, const float* __restrict__ b2,
    float* __restrict__ pairparams)
{
  __shared__ float sb0[64], sb1[64];
  __shared__ float sW2[64 * 5];
  __shared__ float sb2[5];
  __shared__ __align__(16) _Float16 sX[4][16 * 128];
  __shared__ __align__(16) _Float16 sH1[4][16 * 64];
  __shared__ __align__(16) _Float16 sH2[4][16 * 64];
  __shared__ float sOut[4][16 * 5];

  const int tid = threadIdx.x, w = tid >> 5, lane = tid & 31;
  for (int i = tid; i < 64; i += 128) { sb0[i] = b0[i]; sb1[i] = b1[i]; }
  for (int i = tid; i < 320; i += 128) sW2[i] = W2[i];
  if (tid < 5) sb2[tid] = b2[tid];
  __syncthreads();

  const int col = lane & 15, mbase = (lane >> 4) * 8;

  for (int it = 0; it < 4; ++it) {
    int tile = blockIdx.x * 16 + w * 4 + it;
    int pairbase = tile * 16;
    int i1 = pairbase >> 7;        // constant within tile (16 | N)
    int i2b = pairbase & 127;      // contiguous i2 run

    // stage X = [f1(i1) | f2(i2b+r)] (16 x 128 f16), 8-half vector chunks
    for (int cidx = lane; cidx < 256; cidx += 32) {
      int r = cidx >> 4, c8 = (cidx & 15) * 8;
      const _Float16* src = (c8 < 64) ? (f1h + i1 * 64 + c8)
                                      : (f2h + (i2b + r) * 64 + (c8 - 64));
      *(h8*)(sX[w] + r * 128 + c8) = *(const h8*)src;
    }
    for (int idx = lane; idx < 80; idx += 32) sOut[w][idx] = 0.f;
    __syncthreads();

    for (int ord = 0; ord < 2; ++ord) {
      int koff = ord * 64;  // ordering swap = K-rotation of staged X

      // Layer 0: 16x128 @ 128x64
#pragma unroll
      for (int nt = 0; nt < 4; ++nt) {
        v8f acc;
#pragma unroll
        for (int v = 0; v < 8; ++v) acc[v] = sb0[nt * 16 + col];
#pragma unroll
        for (int kt = 0; kt < 4; ++kt) {
          v16h a = load_a_frag(sX[w], 128, kt, koff, 127, lane);
          v16h bb = load_b_packed(W0p, kt * 4 + nt, lane);
          acc = __builtin_amdgcn_wmma_f32_16x16x32_f16(
              false, a, false, bb, (short)0, acc, false, false);
        }
#pragma unroll
        for (int v = 0; v < 8; ++v)
          sH1[w][(mbase + v) * 64 + nt * 16 + col] = (_Float16)milaf(acc[v]);
      }
      __syncthreads();

      // Layer 1: 16x64 @ 64x64
#pragma unroll
      for (int nt = 0; nt < 4; ++nt) {
        v8f acc;
#pragma unroll
        for (int v = 0; v < 8; ++v) acc[v] = sb1[nt * 16 + col];
#pragma unroll
        for (int kt = 0; kt < 2; ++kt) {
          v16h a = load_a_frag(sH1[w], 64, kt, 0, 63, lane);
          v16h bb = load_b_packed(W1p, kt * 4 + nt, lane);
          acc = __builtin_amdgcn_wmma_f32_16x16x32_f16(
              false, a, false, bb, (short)0, acc, false, false);
        }
#pragma unroll
        for (int v = 0; v < 8; ++v)
          sH2[w][(mbase + v) * 64 + nt * 16 + col] = (_Float16)milaf(acc[v]);
      }
      __syncthreads();

      // Layer 2 (N=5, scalar) + ssp, accumulated over orderings
      for (int item = lane; item < 80; item += 32) {
        int r = item / 5, o = item % 5;
        float acc = sb2[o];
#pragma unroll
        for (int k8 = 0; k8 < 64; k8 += 8) {
          h8 hv = *(const h8*)(sH2[w] + r * 64 + k8);
#pragma unroll
          for (int e = 0; e < 8; ++e)
            acc += (float)hv[e] * sW2[(k8 + e) * 5 + o];
        }
        sOut[w][item] += sspf(acc);
      }
      __syncthreads();
    }
    for (int item = lane; item < 80; item += 32)
      pairparams[pairbase * 5 + item] = sOut[w][item];
    __syncthreads();
  }
}

// ---------------------------------------------------------------------------
// block reduce (256 threads)
// ---------------------------------------------------------------------------
__device__ __forceinline__ float block_reduce(float v, float* red) {
  int tid = threadIdx.x;
  red[tid] = v;
  __syncthreads();
  for (int s = 128; s > 0; s >>= 1) {
    if (tid < s) red[tid] += red[tid + s];
    __syncthreads();
  }
  float r = red[0];
  __syncthreads();
  return r;
}

// λ3/λ5 smearing coefficients (SMEAR = -0.39)
__device__ __forceinline__ void pair_coef(float R, float au, float& c3, float& c5) {
  float R2 = R * R, R3 = R2 * R;
  float u3 = R3 / au;
  float exu = __expf(-0.39f * u3);
  float lam3 = 1.f - exu;
  float lam5 = 1.f - (1.f + 0.39f * u3) * exu;
  c3 = lam3 / (R3 + EPSC);
  c5 = 3.f * lam5 / (R2 * R3 + EPSC);
}

// ---------------------------------------------------------------------------
// Induction + pairwise-terms kernel: one block (256 threads) per batch.
// dist (64KB) staged to LDS via async-to-LDS DMA; coords, q, pol, mu, E0 all
// LDS-resident.
// ---------------------------------------------------------------------------
__global__ __launch_bounds__(256) void induction_kernel(
    const float* __restrict__ coords1, const float* __restrict__ coords2,
    const float* __restrict__ dist, const float* __restrict__ multipoles,
    const float* __restrict__ apout, const int* __restrict__ hflags,
    const float* __restrict__ pairparams, float* __restrict__ out)
{
  const int b = blockIdx.x;
  const int tid = threadIdx.x;

  extern __shared__ char smem[];
  float* distS = (float*)smem;            // 16384
  float* c1S = distS + NPAIR;             // 384
  float* c2S = c1S + 384;                 // 384
  float* q1S = c2S + 384;                 // 128
  float* q2S = q1S + 128;                 // 128
  float* sq1S = q2S + 128;                // 128
  float* sq2S = sq1S + 128;               // 128
  float* pol1S = sq2S + 128;              // 128
  float* pol2S = pol1S + 128;             // 128
  float* E01 = pol2S + 128;               // 384
  float* E02 = E01 + 384;                 // 384
  float* m1S = E02 + 384;                 // 384
  float* m2S = m1S + 384;                 // 384
  float* f1h = m2S + 384;                 // 768
  float* f2h = f1h + 768;                 // 768
  float* ad1S = f2h + 768;                // 128 x 8
  float* ad2S = ad1S + 128;
  float* K1S = ad2S + 128;
  float* K2S = K1S + 128;
  float* dq1S = K2S + 128;
  float* dq2S = dq1S + 128;
  float* ae1S = dq2S + 128;
  float* ae2S = ae1S + 128;
  int* h1S = (int*)(ae2S + 128);          // 128
  int* h2S = h1S + 128;                   // 128
  float* red = (float*)(h2S + 128);       // 256

  // prefetch upcoming streams (global_prefetch_b8)
  __builtin_prefetch(pairparams + (size_t)tid * 320, 0, 0);

  // async DMA: dist[b] (64KB) -> LDS, 16B per issue, ASYNCcnt-tracked
  {
    const float* src = dist + (size_t)b * NPAIR;
    for (int chunk = tid; chunk < NPAIR / 4; chunk += 256) {
      unsigned lds_addr = (unsigned)(size_t)(distS + chunk * 4);
      unsigned long long gaddr = (unsigned long long)(size_t)(src + chunk * 4);
      asm volatile("global_load_async_to_lds_b128 %0, %1, off"
                   :: "v"(lds_addr), "v"(gaddr) : "memory");
    }
  }

  for (int idx = tid; idx < 384; idx += 256) {
    c1S[idx] = coords1[(size_t)b * 384 + idx];
    c2S[idx] = coords2[(size_t)b * 384 + idx];
  }
  for (int i = tid; i < NN; i += 256) {
    q1S[i] = multipoles[(size_t)b * NN + i];
    q2S[i] = multipoles[(size_t)NB * NN + (size_t)b * NN + i];
    float p1 = apout[i * 5 + 1], p2 = apout[640 + i * 5 + 1];
    pol1S[i] = p1; pol2S[i] = p2;
    sq1S[i] = sqrtf(p1); sq2S[i] = sqrtf(p2);
    ad1S[i] = apout[i * 5 + 0]; ad2S[i] = apout[640 + i * 5 + 0];
    K1S[i] = apout[i * 5 + 2];  K2S[i] = apout[640 + i * 5 + 2];
    dq1S[i] = apout[i * 5 + 3]; dq2S[i] = apout[640 + i * 5 + 3];
    ae1S[i] = apout[i * 5 + 4]; ae2S[i] = apout[640 + i * 5 + 4];
    h1S[i] = hflags[i]; h2S[i] = hflags[NN + i];
  }
  asm volatile("s_wait_asynccnt 0" ::: "memory");
  __syncthreads();

  const int row = tid >> 1;     // 128 rows x 2 halves
  const int half = tid & 1;

  // --- E0 fields (gather, 2 partials per row) ---
  {
    float ax = 0.f, ay = 0.f, az = 0.f;
    for (int jj = 0; jj < 64; ++jj) {
      int j = half * 64 + jj;
      int p = row * 128 + j;
      float R = distS[p];
      float au = sq1S[row] * sq2S[j] + EPSC;
      float c3, c5; pair_coef(R, au, c3, c5);
      float rx = c1S[row * 3 + 0] - c2S[j * 3 + 0];
      float ry = c1S[row * 3 + 1] - c2S[j * 3 + 1];
      float rz = c1S[row * 3 + 2] - c2S[j * 3 + 2];
      float wq = q2S[j] * c3;
      ax += wq * rx; ay += wq * ry; az += wq * rz;
    }
    f1h[tid * 3 + 0] = ax; f1h[tid * 3 + 1] = ay; f1h[tid * 3 + 2] = az;
  }
  {
    float ax = 0.f, ay = 0.f, az = 0.f;
    int jcol = row;
    for (int ii = 0; ii < 64; ++ii) {
      int i = half * 64 + ii;
      int p = i * 128 + jcol;
      float R = distS[p];
      float au = sq1S[i] * sq2S[jcol] + EPSC;
      float c3, c5; pair_coef(R, au, c3, c5);
      float rx = c1S[i * 3 + 0] - c2S[jcol * 3 + 0];
      float ry = c1S[i * 3 + 1] - c2S[jcol * 3 + 1];
      float rz = c1S[i * 3 + 2] - c2S[jcol * 3 + 2];
      float wq = q1S[i] * c3;
      ax -= wq * rx; ay -= wq * ry; az -= wq * rz;
    }
    f2h[tid * 3 + 0] = ax; f2h[tid * 3 + 1] = ay; f2h[tid * 3 + 2] = az;
  }
  __syncthreads();
  for (int idx = tid; idx < 384; idx += 256) {
    int i = idx / 3, k = idx % 3;
    float e1 = f1h[(2 * i) * 3 + k] + f1h[(2 * i + 1) * 3 + k];
    float e2 = f2h[(2 * i) * 3 + k] + f2h[(2 * i + 1) * 3 + k];
    E01[idx] = e1; E02[idx] = e2;
    m1S[idx] = pol1S[i] * e1;
    m2S[idx] = pol2S[i] * e2;
  }
  __syncthreads();

  // --- 8 Jacobi-damped iterations (T symmetric: T.v = c5 r (r.v) - c3 v) ---
  for (int iter = 0; iter < 8; ++iter) {
    // f1 partials (gather over j)
    {
      float ax = 0.f, ay = 0.f, az = 0.f;
      for (int jj = 0; jj < 64; ++jj) {
        int j = half * 64 + jj;
        int p = row * 128 + j;
        float R = distS[p];
        float au = sq1S[row] * sq2S[j] + EPSC;
        float c3, c5; pair_coef(R, au, c3, c5);
        float rx = c1S[row * 3 + 0] - c2S[j * 3 + 0];
        float ry = c1S[row * 3 + 1] - c2S[j * 3 + 1];
        float rz = c1S[row * 3 + 2] - c2S[j * 3 + 2];
        float vx = m2S[j * 3 + 0], vy = m2S[j * 3 + 1], vz = m2S[j * 3 + 2];
        float dt = rx * vx + ry * vy + rz * vz;
        float c5d = c5 * dt;
        ax += c5d * rx - c3 * vx;
        ay += c5d * ry - c3 * vy;
        az += c5d * rz - c3 * vz;
      }
      f1h[tid * 3 + 0] = ax; f1h[tid * 3 + 1] = ay; f1h[tid * 3 + 2] = az;
    }
    // f2 partials (gather over i)
    {
      float ax = 0.f, ay = 0.f, az = 0.f;
      int jcol = row;
      for (int ii = 0; ii < 64; ++ii) {
        int i = half * 64 + ii;
        int p = i * 128 + jcol;
        float R = distS[p];
        float au = sq1S[i] * sq2S[jcol] + EPSC;
        float c3, c5; pair_coef(R, au, c3, c5);
        float rx = c1S[i * 3 + 0] - c2S[jcol * 3 + 0];
        float ry = c1S[i * 3 + 1] - c2S[jcol * 3 + 1];
        float rz = c1S[i * 3 + 2] - c2S[jcol * 3 + 2];
        float vx = m1S[i * 3 + 0], vy = m1S[i * 3 + 1], vz = m1S[i * 3 + 2];
        float dt = rx * vx + ry * vy + rz * vz;
        float c5d = c5 * dt;
        ax += c5d * rx - c3 * vx;
        ay += c5d * ry - c3 * vy;
        az += c5d * rz - c3 * vz;
      }
      f2h[tid * 3 + 0] = ax; f2h[tid * 3 + 1] = ay; f2h[tid * 3 + 2] = az;
    }
    __syncthreads();
    // mu update: m = 0.25 m + 0.75 p (E0 + f)
    for (int idx = tid; idx < 384; idx += 256) {
      int i = idx / 3, k = idx % 3;
      float s1 = E01[idx] + f1h[(2 * i) * 3 + k] + f1h[(2 * i + 1) * 3 + k];
      float s2 = E02[idx] + f2h[(2 * i) * 3 + k] + f2h[(2 * i + 1) * 3 + k];
      m1S[idx] = 0.25f * m1S[idx] + 0.75f * pol1S[i] * s1;
      m2S[idx] = 0.25f * m2S[idx] + 0.75f * pol2S[i] * s2;
    }
    __syncthreads();
  }

  // induction energy
  float eloc = 0.f;
  for (int idx = tid; idx < 384; idx += 256)
    eloc += m1S[idx] * E01[idx] + m2S[idx] * E02[idx];
  float e_ind = -0.5f * block_reduce(eloc, red);

  // --- pairwise terms (es, ct, ex, disp) ---
  float es_s = 0.f, ct_s = 0.f, ex_s = 0.f, disp_s = 0.f;
  for (int p = tid; p < NPAIR; p += 256) {
    int i = p >> 7, j = p & 127;
    float R = distS[p];
    float B0 = 1.f / (R + EPSC);
    float R2 = R * R;
    const float* pp = pairparams + (size_t)p * 5;
    float C6 = pp[0], C8 = pp[1], C10 = pp[2], Act = pp[3], bct = pp[4];
    // charge-transfer correction (subtracted from induction)
    ct_s += Act * __expf(-bct * R);
    // dispersion (BJ-damped), cutoff R < 50
    {
      float r02 = C8 / (C6 + EPSC);
      float r3 = r02 * r02 * r02, r4 = r3 * r02, r5 = r4 * r02;
      float R6 = R2 * R2 * R2, R8 = R6 * R2, R10 = R8 * R2;
      float bj = -(C6 / (R6 + r3) + C8 / (R8 + r4) + C10 / (R10 + r5));
      if (R2 < 2500.f) disp_s += bj;
    }
    // electrostatics
    {
      float bd = sqrtf(ad1S[i] * ad2S[j]);
      float ebd = __expf(-bd * R);
      float fd = 1.f - (1.f + 0.5f * bd * R) * ebd;
      es_s += q1S[i] * q2S[j] * fd * B0;
    }
    // exchange
    {
      float be = sqrtf(ae1S[i] * ae2S[j]);
      float Kp = K1S[i] * K2S[j];
      float qe1 = h1S[i] ? (1.f - q1S[i]) : (2.f + dq1S[i] - q1S[i]);
      float qe2 = h2S[j] ? (1.f - q2S[j]) : (2.f + dq2S[j] - q2S[j]);
      float rx = c2S[j * 3 + 0] - c1S[i * 3 + 0];  // Rx1 = c2 - c1
      float ry = c2S[j * 3 + 1] - c1S[i * 3 + 1];
      float rz = c2S[j * 3 + 2] - c1S[i * 3 + 2];
      float proj = ((m1S[i * 3 + 0] - m2S[j * 3 + 0]) * rx +
                    (m1S[i * 3 + 1] - m2S[j * 3 + 1]) * ry +
                    (m1S[i * 3 + 2] - m2S[j * 3 + 2]) * rz) * B0;
      ex_s += Kp * qe1 * qe2 * (1.f + proj) * __expf(-be * R) * B0;
    }
  }
  float es_t = block_reduce(es_s, red);
  float ct_t = block_reduce(ct_s, red);
  float ex_t = block_reduce(ex_s, red);
  float disp_t = block_reduce(disp_s, red);

  if (tid == 0) {
    float in_t = e_ind - ct_t;
    out[0 * NB + b] = ex_t + disp_t + es_t + in_t;  // total
    out[1 * NB + b] = es_t;                          // es_term
    out[2 * NB + b] = in_t;                          // in_term
    out[3 * NB + b] = ex_t;                          // ex_term
    out[4 * NB + b] = disp_t;                        // disp_term
  }
}

// ---------------------------------------------------------------------------
// Launcher
// ---------------------------------------------------------------------------
extern "C" void kernel_launch(void* const* d_in, const int* in_sizes, int n_in,
                              void* d_out, int out_size, void* d_ws, size_t ws_size,
                              hipStream_t stream) {
  (void)in_sizes; (void)n_in; (void)out_size; (void)ws_size;

  const float* nodes1 = (const float*)d_in[0];
  const float* edges1 = (const float*)d_in[1];
  const int* senders1 = (const int*)d_in[2];
  const int* receivers1 = (const int*)d_in[3];
  const float* nodes2 = (const float*)d_in[4];
  const float* edges2 = (const float*)d_in[5];
  const int* senders2 = (const int*)d_in[6];
  const int* receivers2 = (const int*)d_in[7];
  const float* coords1 = (const float*)d_in[8];
  const float* coords2 = (const float*)d_in[9];
  const float* distm = (const float*)d_in[10];
  const float* multipoles = (const float*)d_in[11];
  const float* W_emb_node = (const float*)d_in[12];
  const float* b_emb_node = (const float*)d_in[13];
  const float* W_emb_edge = (const float*)d_in[14];
  const float* b_emb_edge = (const float*)d_in[15];
  const float* W_gn_edge = (const float*)d_in[16];
  const float* b_gn_edge = (const float*)d_in[17];
  const float* W_gn_node = (const float*)d_in[18];
  const float* b_gn_node = (const float*)d_in[19];
  const float* pp_W0 = (const float*)d_in[20];
  const float* pp_b0 = (const float*)d_in[21];
  const float* pp_W1 = (const float*)d_in[22];
  const float* pp_b1 = (const float*)d_in[23];
  const float* pp_W2 = (const float*)d_in[24];
  const float* pp_b2 = (const float*)d_in[25];
  const float* ap_W0 = (const float*)d_in[26];
  const float* ap_b0 = (const float*)d_in[27];
  const float* ap_W1 = (const float*)d_in[28];
  const float* ap_b1 = (const float*)d_in[29];
  const float* ap_W2 = (const float*)d_in[30];
  const float* ap_b2 = (const float*)d_in[31];

  char* ws = (char*)d_ws;
  float*    feats_f32  = (float*)(ws + 0);           // 2*8192 f32  = 65536 B
  _Float16* feats_f16  = (_Float16*)(ws + 65536);    // 2*8192 f16  = 32768 B
  float*    apout      = (float*)(ws + 98304);       // 2*640 f32   = 5120 B
  int*      hflags     = (int*)(ws + 103424);        // 2*128 i32   = 1024 B
  float*    e_buf      = (float*)(ws + 104448);      // 2*32768 f32 = 262144 B
  _Float16* W0p        = (_Float16*)(ws + 366592);   // 8192 f16    = 16384 B
  _Float16* W1p        = (_Float16*)(ws + 382976);   // 4096 f16    = 8192 B
  float*    pairparams = (float*)(ws + 391168);      // 16384*5 f32 = 327680 B

  convert_pack_kernel<<<32, 256, 0, stream>>>(pp_W0, pp_W1, W0p, W1p);

  gnn_kernel<<<2, 256, 2 * NN * HS * sizeof(float), stream>>>(
      nodes1, edges1, senders1, receivers1,
      nodes2, edges2, senders2, receivers2,
      W_emb_node, b_emb_node, W_emb_edge, b_emb_edge,
      W_gn_edge, b_gn_edge, W_gn_node, b_gn_node,
      ap_W0, ap_b0, ap_W1, ap_b1, ap_W2, ap_b2,
      feats_f32, feats_f16, apout, hflags, e_buf);

  pair_kernel<<<64, 128, 0, stream>>>(
      feats_f16, feats_f16 + NN * HS, W0p, W1p,
      pp_b0, pp_b1, pp_W2, pp_b2, pairparams);

  size_t ind_smem = (size_t)22528 * sizeof(float);   // 90112 B (fits 320KB WGP LDS)
  induction_kernel<<<NB, 256, ind_smem, stream>>>(
      coords1, coords2, distm, multipoles, apout, hflags,
      pairparams, (float*)d_out);
}
